// DecoderLayerLSTM_36395552866595
// MI455X (gfx1250) — matrix operand order repaired and checked by reading, not compiled, not run on hardware
//
#include <hip/hip_runtime.h>
#include <hip/hip_bf16.h>

// Problem constants (from reference): V=32000, E=512, H=1024, L=2, B=64
#define CV 32000
#define CE 512
#define CH 1024
#define CB 64

typedef __attribute__((ext_vector_type(16))) _Float16 v16h;
typedef __attribute__((ext_vector_type(8)))  float    v8f;

__device__ __forceinline__ float4 ld4(const float* p) {
    return *(const float4*)p;
}

__device__ __forceinline__ v16h pack16(float4 a, float4 b, float4 c, float4 d) {
    v16h v;
    v[0]  = (_Float16)a.x; v[1]  = (_Float16)a.y; v[2]  = (_Float16)a.z; v[3]  = (_Float16)a.w;
    v[4]  = (_Float16)b.x; v[5]  = (_Float16)b.y; v[6]  = (_Float16)b.z; v[7]  = (_Float16)b.w;
    v[8]  = (_Float16)c.x; v[9]  = (_Float16)c.y; v[10] = (_Float16)c.z; v[11] = (_Float16)c.w;
    v[12] = (_Float16)d.x; v[13] = (_Float16)d.y; v[14] = (_Float16)d.z; v[15] = (_Float16)d.w;
    return v;
}

// A-matrix fragment (16x32 f16): lane holds row m, halves 0..7 at p[0..7], 8..15 at p[16..23]
__device__ __forceinline__ v16h packA(const float* p) {
    return pack16(ld4(p), ld4(p + 4), ld4(p + 16), ld4(p + 20));
}

// B-matrix fragment (32x16 f16): lane holds column n, halves 0..7 at p[0..7], 8..15 at p[16..23]
__device__ __forceinline__ v16h packB(const float* p) {
    return pack16(ld4(p), ld4(p + 4), ld4(p + 16), ld4(p + 20));
}

__device__ __forceinline__ float sigm(float x) {
    return 1.0f / (1.0f + __expf(-x));
}

// ---------------------------------------------------------------------------
// Embedding gather: x[m][e] = emb_w[idx[m]][e]    (B*E = 32768 elements)
// ---------------------------------------------------------------------------
__global__ __launch_bounds__(256) void embed_kernel(
        const int* __restrict__ idx, const float* __restrict__ emb,
        float* __restrict__ x) {
    int i = blockIdx.x * 256 + threadIdx.x;     // 0 .. B*E-1
    int m = i >> 9;                             // /E
    int e = i & (CE - 1);
    x[i] = emb[(size_t)idx[m] * CE + e];
}

// ---------------------------------------------------------------------------
// One LSTM cell, fully fused:
//   gates = xin @ w_ih.T + h_prev @ w_hh.T + b_ih + b_hh ; pointwise -> h,c
// Grid: 64 hidden-chunks(16) x 2 batch-halves(32) = 128 WGs, 256 threads.
// Wave w: gate g = w&3, m-tile mt = w>>2 ; one 16x16 WMMA tile per wave.
// ---------------------------------------------------------------------------
__global__ __launch_bounds__(256) void lstm_layer_kernel(
        const float* __restrict__ xin, int Kx,
        const float* __restrict__ w_ih, const float* __restrict__ w_hh,
        const float* __restrict__ b_ih, const float* __restrict__ b_hh,
        const float* __restrict__ h_prev, const float* __restrict__ c_prev,
        float* __restrict__ h_out, float* __restrict__ c_out) {
    __shared__ float gbuf[4][32][16];           // [gate][m_local][j]

    const int jc   = blockIdx.x >> 1;
    const int m0   = (blockIdx.x & 1) * 32;
    const int j0   = jc * 16;
    const int tid  = threadIdx.x;
    const int wave = tid >> 5;
    const int lane = tid & 31;
    const int g    = wave & 3;
    const int mt   = wave >> 2;                 // 0 or 1

    const int mrow = m0 + mt * 16 + (lane & 15);        // A row for this lane
    const int n    = g * CH + j0 + (lane & 15);         // gate column (0..4H)
    const int ha   = (lane >> 4) << 3;                  // A lane-half K offset (0/8)
    const int hb   = (lane >> 4) << 3;                  // B lane-half K offset (0/8)

    v8f acc = {};

    // ---- phase 1: x @ w_ih.T ----
    {
        const float* wr = w_ih + (size_t)n * Kx;
        const float* ar = xin  + (size_t)mrow * Kx;
        for (int k0 = 0; k0 < Kx; k0 += 32) {
            __builtin_prefetch(wr + k0 + 256, 0, 1);    // global_prefetch_b8
            v16h a = packA(ar + k0 + ha);
            v16h b = packB(wr + k0 + hb);
            acc = __builtin_amdgcn_wmma_f32_16x16x32_f16(
                      false, a, false, b, (short)0, acc, false, false);
        }
    }
    // ---- phase 2: h_prev @ w_hh.T ----
    {
        const float* wr = w_hh   + (size_t)n * CH;
        const float* ar = h_prev + (size_t)mrow * CH;
        for (int k0 = 0; k0 < CH; k0 += 32) {
            __builtin_prefetch(wr + k0 + 256, 0, 1);
            v16h a = packA(ar + k0 + ha);
            v16h b = packB(wr + k0 + hb);
            acc = __builtin_amdgcn_wmma_f32_16x16x32_f16(
                      false, a, false, b, (short)0, acc, false, false);
        }
    }

    // ---- scatter C tile into LDS (C/D layout: VGPR r -> rows r / r+8) ----
    const int mlb = mt * 16 + ha;               // local row base
    const int jl  = lane & 15;
#pragma unroll
    for (int r = 0; r < 8; ++r)
        gbuf[g][mlb + r][jl] = acc[r];
    __syncthreads();

    // ---- fused LSTM pointwise: 32x16 = 512 outputs, 2 per thread ----
#pragma unroll
    for (int p = 0; p < 2; ++p) {
        int i  = tid + p * 256;
        int ml = i >> 4;
        int j  = i & 15;
        int m  = m0 + ml;
        int jg = j0 + j;
        float iv = gbuf[0][ml][j] + b_ih[0 * CH + jg] + b_hh[0 * CH + jg];
        float fv = gbuf[1][ml][j] + b_ih[1 * CH + jg] + b_hh[1 * CH + jg];
        float gv = gbuf[2][ml][j] + b_ih[2 * CH + jg] + b_hh[2 * CH + jg];
        float ov = gbuf[3][ml][j] + b_ih[3 * CH + jg] + b_hh[3 * CH + jg];
        float ig = sigm(iv);
        float fg = sigm(fv);
        float gg = tanhf(gv);
        float og = sigm(ov);
        float c  = c_prev[(size_t)m * CH + jg];
        float cn = fg * c + ig * gg;
        float hn = og * tanhf(cn);
        h_out[(size_t)m * CH + jg] = hn;
        c_out[(size_t)m * CH + jg] = cn;
    }
}

// ---------------------------------------------------------------------------
// FC: logit[64, 32000] = h1 @ fc_w.T + fc_b
// Grid: 500 WGs x 64 columns; wave: n-tile = w&3, two M-tiles (B-frag reuse).
// ---------------------------------------------------------------------------
__global__ __launch_bounds__(256) void fc_kernel(
        const float* __restrict__ h1, const float* __restrict__ fc_w,
        const float* __restrict__ fc_b, float* __restrict__ out) {
    const int n0   = blockIdx.x * 64;
    const int wave = threadIdx.x >> 5;
    const int lane = threadIdx.x & 31;
    const int nt   = wave & 3;
    const int mb   = (wave >> 2) * 32;
    const int n    = n0 + nt * 16 + (lane & 15);
    const int ha   = (lane >> 4) << 3;
    const int hb   = (lane >> 4) << 3;

    const float* wr  = fc_w + (size_t)n * CH;
    const float* ar0 = h1 + (size_t)(mb      + (lane & 15)) * CH;
    const float* ar1 = h1 + (size_t)(mb + 16 + (lane & 15)) * CH;

    v8f acc0 = {};
    v8f acc1 = {};
    for (int k0 = 0; k0 < CH; k0 += 32) {
        __builtin_prefetch(wr + k0 + 256, 0, 1);
        v16h b  = packB(wr  + k0 + hb);
        v16h a0 = packA(ar0 + k0 + ha);
        v16h a1 = packA(ar1 + k0 + ha);
        acc0 = __builtin_amdgcn_wmma_f32_16x16x32_f16(
                   false, a0, false, b, (short)0, acc0, false, false);
        acc1 = __builtin_amdgcn_wmma_f32_16x16x32_f16(
                   false, a1, false, b, (short)0, acc1, false, false);
    }

    const float bias = fc_b[n];
#pragma unroll
    for (int r = 0; r < 8; ++r) {
        int ma = mb + r + ha;
        out[(size_t)ma * CV + n] = acc0[r] + bias;
        int mc = mb + 16 + r + ha;
        out[(size_t)mc * CV + n] = acc1[r] + bias;
    }
}

// ---------------------------------------------------------------------------
// Launch
// ---------------------------------------------------------------------------
extern "C" void kernel_launch(void* const* d_in, const int* in_sizes, int n_in,
                              void* d_out, int out_size, void* d_ws, size_t ws_size,
                              hipStream_t stream) {
    const int*   idx     = (const int*)  d_in[0];
    const float* h_state = (const float*)d_in[1];   // [L,B,H]
    const float* c_state = (const float*)d_in[2];   // [L,B,H]
    const float* emb_w   = (const float*)d_in[3];   // [V,E]
    const float* w_ih0   = (const float*)d_in[4];   // [4H,E]
    const float* w_hh0   = (const float*)d_in[5];   // [4H,H]
    const float* b_ih0   = (const float*)d_in[6];
    const float* b_hh0   = (const float*)d_in[7];
    const float* w_ih1   = (const float*)d_in[8];   // [4H,H]
    const float* w_hh1   = (const float*)d_in[9];
    const float* b_ih1   = (const float*)d_in[10];
    const float* b_hh1   = (const float*)d_in[11];
    const float* fc_w    = (const float*)d_in[12];  // [V,H]
    const float* fc_b    = (const float*)d_in[13];

    float* out   = (float*)d_out;
    float* logit = out;                              // [B,V]
    float* h_new = out + (size_t)CB * CV;            // [L,B,H]
    float* c_new = h_new + (size_t)2 * CB * CH;      // [L,B,H]

    float* x = (float*)d_ws;                         // [B,E] gathered embeddings

    // 1) embedding gather
    embed_kernel<<<(CB * CE) / 256, 256, 0, stream>>>(idx, emb_w, x);

    // 2) LSTM layer 0: x[B,E] -> h_new[0], c_new[0]
    lstm_layer_kernel<<<128, 256, 0, stream>>>(
        x, CE, w_ih0, w_hh0, b_ih0, b_hh0,
        h_state, c_state, h_new, c_new);

    // 3) LSTM layer 1: h0_new[B,H] -> h_new[1], c_new[1]
    lstm_layer_kernel<<<128, 256, 0, stream>>>(
        h_new, CH, w_ih1, w_hh1, b_ih1, b_hh1,
        h_state + (size_t)CB * CH, c_state + (size_t)CB * CH,
        h_new + (size_t)CB * CH, c_new + (size_t)CB * CH);

    // 4) FC: h1_new -> logits
    fc_kernel<<<CV / 64, 256, 0, stream>>>(
        h_new + (size_t)CB * CH, fc_w, fc_b, logit);
}
